// GraphRNN_64123861729292
// MI455X (gfx1250) — matrix operand order repaired and checked by reading, not compiled, not run on hardware
//
#include <hip/hip_runtime.h>

typedef __attribute__((ext_vector_type(16))) __bf16 v16bf;
typedef __attribute__((ext_vector_type(8)))  float  v8f;

#define D_DIM   512
#define G_DIM   1536          // 3*D
#define N_NODES 207
#define B_BATCH 16
#define T_STEPS 24
#define L_LAYERS 2
#define M_ROWS  (B_BATCH * N_NODES)   // 3312
#define KTILES  (D_DIM / 32)          // 16
#define NTILES  (G_DIM / 16)          // 96
#define NB_PER_WAVE 8                 // 8 x 16 = 128 columns per wave
#define LN_EPS  1e-5f

// ---------------------------------------------------------------------------
// Pack fp32 weights [L][D][3D] into bf16 WMMA B-fragment layout:
//   packed[l][w][ntile][ktile][lane(32)][j(16)]
//   lane = 16*khalf + n  (n = column within 16-wide tile)
//   element j -> K = ktile*32 + 16*khalf + j
// ---------------------------------------------------------------------------
__global__ __launch_bounds__(256)
void pack_weights_kernel(const float* __restrict__ Wi,
                         const float* __restrict__ Wh,
                         __bf16* __restrict__ packed) {
    size_t idx = (size_t)blockIdx.x * blockDim.x + threadIdx.x;
    const size_t total = (size_t)L_LAYERS * 2 * NTILES * KTILES * 512;
    if (idx >= total) return;
    int j    = (int)(idx & 15);
    int lane = (int)((idx >> 4) & 31);
    int kt   = (int)((idx >> 9) % KTILES);
    int nt   = (int)((idx / (512u * KTILES)) % NTILES);
    int w    = (int)((idx / ((size_t)512 * KTILES * NTILES)) & 1);
    int l    = (int)( idx / ((size_t)512 * KTILES * NTILES * 2));
    int n = nt * 16 + (lane & 15);
    int k = kt * 32 + ((lane >> 4) << 4) + j;
    const float* W = (w == 0 ? Wi : Wh) + (size_t)l * D_DIM * G_DIM;
    packed[idx] = (__bf16)W[(size_t)k * G_DIM + n];
}

// ---------------------------------------------------------------------------
// h0 [B, L, N, D] -> h_buf [L, B*N, D]
// ---------------------------------------------------------------------------
__global__ __launch_bounds__(256)
void init_h_kernel(const float* __restrict__ h0, float* __restrict__ hbuf) {
    size_t idx = (size_t)blockIdx.x * blockDim.x + threadIdx.x;
    const size_t total = (size_t)B_BATCH * L_LAYERS * N_NODES * D_DIM;
    if (idx >= total) return;
    int d = (int)(idx % D_DIM);
    int n = (int)((idx / D_DIM) % N_NODES);
    int l = (int)((idx / ((size_t)D_DIM * N_NODES)) % L_LAYERS);
    int b = (int)( idx / ((size_t)D_DIM * N_NODES * L_LAYERS));
    hbuf[((size_t)l * M_ROWS + (size_t)b * N_NODES + n) * D_DIM + d] =
        h0[(((size_t)b * L_LAYERS + l) * N_NODES + n) * D_DIM + d];
}

// ---------------------------------------------------------------------------
// h_buf [L, B*N, D] -> out hidden section [B, L, N, D]
// ---------------------------------------------------------------------------
__global__ __launch_bounds__(256)
void final_h_kernel(const float* __restrict__ hbuf, float* __restrict__ outh) {
    size_t idx = (size_t)blockIdx.x * blockDim.x + threadIdx.x;
    const size_t total = (size_t)B_BATCH * L_LAYERS * N_NODES * D_DIM;
    if (idx >= total) return;
    int d = (int)(idx % D_DIM);
    int n = (int)((idx / D_DIM) % N_NODES);
    int l = (int)((idx / ((size_t)D_DIM * N_NODES)) % L_LAYERS);
    int b = (int)( idx / ((size_t)D_DIM * N_NODES * L_LAYERS));
    outh[idx] = hbuf[((size_t)l * M_ROWS + (size_t)b * N_NODES + n) * D_DIM + d];
}

// ---------------------------------------------------------------------------
// GEMM: dst[M_ROWS, G_DIM] = srcA[M_ROWS, D] @ W(packed bf16) + bias
// 4 waves/block; each wave: 16(M) x 128(N) tile -> 8 WMMA per K-step of 32.
// x_layout=1: srcA rows live at ((b*T + t)*N + n)*D  (b = r/N, n = r%N)
// x_layout=0: srcA rows contiguous at r*D
// ---------------------------------------------------------------------------
__global__ __launch_bounds__(128)
void gemm_bf16_kernel(const float* __restrict__ srcA,
                      const __bf16* __restrict__ wpack,
                      const float* __restrict__ bias,
                      float* __restrict__ dst,
                      int t, int x_layout) {
    const int wave  = threadIdx.x >> 5;                     // 0..3
    const int lane  = threadIdx.x & 31;
    const int m0    = blockIdx.x * 16;                      // 207 blocks along M
    const int n0    = blockIdx.y * (NB_PER_WAVE * 16 * 4) + wave * (NB_PER_WAVE * 16);
    const int khalf = lane >> 4;

    // This lane's activation row (A layout: lanes 0-15 & 16-31 both hold M=0..15)
    const int mrow = m0 + (lane & 15);
    size_t row_base;
    if (x_layout) {
        int b = mrow / N_NODES;
        int n = mrow - b * N_NODES;
        row_base = (((size_t)b * T_STEPS + t) * N_NODES + n) * D_DIM;
    } else {
        row_base = (size_t)mrow * D_DIM;
    }

    v8f acc[NB_PER_WAVE];
#pragma unroll
    for (int nb = 0; nb < NB_PER_WAVE; ++nb) acc[nb] = v8f{};

    const int nt0 = n0 >> 4;                                // first 16-wide column tile
    for (int kt = 0; kt < KTILES; ++kt) {
        // A fragment: element pair (2v, 2v+1) -> K = kt*32 + 16*khalf + 2*v
        v16bf a;
#pragma unroll
        for (int v = 0; v < 8; ++v) {
            int k = kt * 32 + (khalf << 4) + (v << 1);
            float f0 = srcA[row_base + k];
            float f1 = srcA[row_base + k + 1];
            a[2 * v]     = (__bf16)f0;
            a[2 * v + 1] = (__bf16)f1;
        }
#pragma unroll
        for (int nb = 0; nb < NB_PER_WAVE; ++nb) {
            const v16bf* bp = (const v16bf*)(wpack +
                (((size_t)(nt0 + nb) * KTILES + kt) << 9) + (lane << 4));
            v16bf bfrag = *bp;
            acc[nb] = __builtin_amdgcn_wmma_f32_16x16x32_bf16(
                false, a, false, bfrag, (short)0, acc[nb], false, false);
        }
    }

    // Epilogue: C/D layout -> row = m0 + 8*(lane>>4) + i, col = lane&15
    const int coll = lane & 15;
#pragma unroll
    for (int nb = 0; nb < NB_PER_WAVE; ++nb) {
        int col = n0 + nb * 16 + coll;
        float bv = bias[col];
#pragma unroll
        for (int i = 0; i < 8; ++i) {
            int row = m0 + (khalf << 3) + i;
            dst[(size_t)row * G_DIM + col] = acc[nb][i] + bv;
        }
    }
}

// ---------------------------------------------------------------------------
// Fused GRU gates + LayerNorm. One block (256 thr) per row; 2 elems/thread.
// ---------------------------------------------------------------------------
__global__ __launch_bounds__(256)
void gru_elem_ln_kernel(const float* __restrict__ gi,
                        const float* __restrict__ gh,
                        float* __restrict__ h,            // [M_ROWS, D] this layer
                        const float* __restrict__ gamma,
                        const float* __restrict__ beta,
                        float* __restrict__ out_slice,    // null unless top layer
                        int t) {
    __shared__ float ssum[256];
    __shared__ float ssq[256];
    const int r   = blockIdx.x;
    const int tid = threadIdx.x;
    const size_t gbase = (size_t)r * G_DIM;
    const size_t hbase = (size_t)r * D_DIM;

    float vals[2];
    float s = 0.f, s2 = 0.f;
#pragma unroll
    for (int q = 0; q < 2; ++q) {
        int d = tid + q * 256;
        float ir = gi[gbase + d];
        float iz = gi[gbase + D_DIM + d];
        float in_ = gi[gbase + 2 * D_DIM + d];
        float hr = gh[gbase + d];
        float hz = gh[gbase + D_DIM + d];
        float hn = gh[gbase + 2 * D_DIM + d];
        float hp = h[hbase + d];
        float gr = 1.f / (1.f + __expf(-(ir + hr)));
        float gz = 1.f / (1.f + __expf(-(iz + hz)));
        float nn = tanhf(in_ + gr * hn);
        float o  = nn + gz * (hp - nn);
        vals[q] = o;
        s += o;
        s2 += o * o;
    }
    ssum[tid] = s;
    ssq[tid]  = s2;
    __syncthreads();
    for (int w = 128; w > 0; w >>= 1) {
        if (tid < w) { ssum[tid] += ssum[tid + w]; ssq[tid] += ssq[tid + w]; }
        __syncthreads();
    }
    const float mu   = ssum[0] * (1.f / D_DIM);
    const float var  = ssq[0] * (1.f / D_DIM) - mu * mu;
    const float rstd = rsqrtf(var + LN_EPS);

    size_t obase = 0;
    if (out_slice) {
        int b = r / N_NODES;
        int n = r - b * N_NODES;
        obase = (((size_t)b * T_STEPS + t) * N_NODES + n) * D_DIM;
    }
#pragma unroll
    for (int q = 0; q < 2; ++q) {
        int d = tid + q * 256;
        float y = (vals[q] - mu) * rstd * gamma[d] + beta[d];
        h[hbase + d] = y;
        if (out_slice) out_slice[obase + d] = y;
    }
}

// ---------------------------------------------------------------------------
extern "C" void kernel_launch(void* const* d_in, const int* in_sizes, int n_in,
                              void* d_out, int out_size, void* d_ws, size_t ws_size,
                              hipStream_t stream) {
    (void)in_sizes; (void)n_in; (void)out_size; (void)ws_size;
    const float* x     = (const float*)d_in[0];
    const float* h0    = (const float*)d_in[1];
    const float* Wi    = (const float*)d_in[2];
    const float* bi    = (const float*)d_in[3];
    const float* Wh    = (const float*)d_in[4];
    const float* bh    = (const float*)d_in[5];
    const float* gamma = (const float*)d_in[6];
    const float* beta  = (const float*)d_in[7];
    float* out = (float*)d_out;

    // workspace carve (256B aligned)
    char* ws = (char*)d_ws;
    size_t off = 0;
    auto carve = [&](size_t bytes) -> void* {
        void* p = ws + off;
        off = (off + bytes + 255) & ~(size_t)255;
        return p;
    };
    const size_t wElems = (size_t)D_DIM * G_DIM;      // per (l,w) weight
    __bf16* wpack = (__bf16*)carve((size_t)L_LAYERS * 2 * wElems * sizeof(__bf16));
    float*  gibuf = (float*)carve((size_t)M_ROWS * G_DIM * sizeof(float));
    float*  ghbuf = (float*)carve((size_t)M_ROWS * G_DIM * sizeof(float));
    float*  hbuf  = (float*)carve((size_t)L_LAYERS * M_ROWS * D_DIM * sizeof(float));

    // one-time prep
    {
        size_t total = (size_t)L_LAYERS * 2 * NTILES * KTILES * 512;
        pack_weights_kernel<<<dim3((unsigned)((total + 255) / 256)), 256, 0, stream>>>(Wi, Wh, wpack);
        size_t th = (size_t)B_BATCH * L_LAYERS * N_NODES * D_DIM;
        init_h_kernel<<<dim3((unsigned)((th + 255) / 256)), 256, 0, stream>>>(h0, hbuf);
    }

    const dim3 gemm_grid(M_ROWS / 16, G_DIM / (NB_PER_WAVE * 16 * 4));   // 207 x 3
    for (int t = 0; t < T_STEPS; ++t) {
        for (int l = 0; l < L_LAYERS; ++l) {
            const float* srcx = (l == 0) ? x : (hbuf /* layer 0 state */);
            float* hl = hbuf + (size_t)l * M_ROWS * D_DIM;
            // gi = x_in @ Wi[l] + bi[l]
            gemm_bf16_kernel<<<gemm_grid, 128, 0, stream>>>(
                srcx, wpack + ((size_t)l * 2 + 0) * wElems, bi + (size_t)l * G_DIM,
                gibuf, t, (l == 0) ? 1 : 0);
            // gh = h[l] @ Wh[l] + bh[l]
            gemm_bf16_kernel<<<gemm_grid, 128, 0, stream>>>(
                hl, wpack + ((size_t)l * 2 + 1) * wElems, bh + (size_t)l * G_DIM,
                ghbuf, t, 0);
            // gates + LN; top layer also writes the output sequence
            gru_elem_ln_kernel<<<dim3(M_ROWS), 256, 0, stream>>>(
                gibuf, ghbuf, hl, gamma + (size_t)l * D_DIM, beta + (size_t)l * D_DIM,
                (l == L_LAYERS - 1) ? out : nullptr, t);
        }
    }

    // hidden state -> tail of d_out
    {
        float* outh = out + (size_t)B_BATCH * T_STEPS * N_NODES * D_DIM;
        size_t th = (size_t)B_BATCH * L_LAYERS * N_NODES * D_DIM;
        final_h_kernel<<<dim3((unsigned)((th + 255) / 256)), 256, 0, stream>>>(hbuf, outh);
    }
}
